// GPT_67877663146205
// MI455X (gfx1250) — compile-verified, matrix operand order checked by reading, not run on hardware
//
#include <hip/hip_runtime.h>

// GPT forward: embed -> QKV -> causal flash attention -> LM head.
// B=2 T=2048 V=32000 E=1024 H=16 D=64.
// GEMMs: v_wmma_f32_16x16x32_f16.  Tile staging: global_load_async_to_lds_b128
// (ASYNCcnt) from pre-swizzled f16 weight/K/V tiles, double buffered.

#define B_ 2
#define T_ 2048
#define V_ 32000
#define E_ 1024
#define H_ 16
#define D_ 64
#define BT_ (B_ * T_)
#define KT_ (E_ / 32)          /* 32 K-tiles over E */

typedef _Float16 h16;
typedef __attribute__((ext_vector_type(16))) _Float16 v16h;
typedef __attribute__((ext_vector_type(8)))  _Float16 v8h;
typedef __attribute__((ext_vector_type(8)))  float    v8f;
typedef __attribute__((ext_vector_type(4)))  float    v4f;
typedef __attribute__((ext_vector_type(4)))  int      v4i;

union Frag16 {
  v16h v;
  v8h  h2[2];
  h16  e[16];
};

__device__ __forceinline__ v8f wmma16(const Frag16& a, const Frag16& b, v8f c) {
  return __builtin_amdgcn_wmma_f32_16x16x32_f16(false, a.v, false, b.v,
                                                (short)0, c, false, false);
}

// B-fragment swizzle for a 32(K) x 64(N) f16 tile: lane L of fragment nt reads
// its 16 halves contiguously at halves[(nt*32 + L)*16].  Fragment half h of
// lane (g=L/16, n=L%16) holds B[k][n] with k = h + (h&8) + 8*g.
__device__ __forceinline__ int swzB(int k, int n) {
  int g = (k >> 3) & 1;
  int h = (k & 7) | ((k >> 4) << 3);
  return (((n >> 4) * 32 + g * 16 + (n & 15)) << 4) + h;
}

// ---------------- async global->LDS 16B copy + asynccnt waits --------------
#ifndef __has_builtin
#define __has_builtin(x) 0
#endif

__device__ __forceinline__ void async_cp16(h16* ldst, const h16* gsrc) {
#if __has_builtin(__builtin_amdgcn_global_load_async_to_lds_b128)
  // signature (from hipcc diagnostic): (int4 AS1*, int4 AS3*, imm, imm)
  __builtin_amdgcn_global_load_async_to_lds_b128(
      (__attribute__((address_space(1))) v4i*)gsrc,
      (__attribute__((address_space(3))) v4i*)ldst, 0, 0);
#else
  unsigned loff =
      (unsigned)(size_t)(__attribute__((address_space(3))) h16*)ldst;
  asm volatile("global_load_async_to_lds_b128 %0, %1, off"
               :: "v"(loff), "v"(gsrc) : "memory");
#endif
}

#if __has_builtin(__builtin_amdgcn_s_wait_asynccnt)
#define WAIT_ASYNC(n) __builtin_amdgcn_s_wait_asynccnt(n)
#else
#define WAIT_ASYNC(n) asm volatile("s_wait_asynccnt %0" :: "i"(n) : "memory")
#endif

// ---------------------------------------------------------------- embed
__global__ __launch_bounds__(256) void k_embed(const int* __restrict__ X,
                                               const float* __restrict__ tok,
                                               const float* __restrict__ pos,
                                               h16* __restrict__ x16) {
  const int row = blockIdx.x;
  const int t   = row & (T_ - 1);
  const int vtk = X[row];
  const int e   = threadIdx.x * 4;
  v4f a = *(const v4f*)(tok + (size_t)vtk * E_ + e);
  v4f p = *(const v4f*)(pos + (size_t)t * E_ + e);
  a = a + p;
  h16* o = x16 + (size_t)row * E_ + e;
  o[0] = (h16)a[0]; o[1] = (h16)a[1]; o[2] = (h16)a[2]; o[3] = (h16)a[3];
}

// ------------------------------------------- pre-swizzle QKV weights (f16)
// grid: (KT_, H).  One 32x64 fragment-ordered tile per block per matrix.
__global__ __launch_bounds__(256) void k_prepqkv(const float* __restrict__ Wq,
                                                 const float* __restrict__ Wk,
                                                 const float* __restrict__ Wv,
                                                 h16* __restrict__ wq16,
                                                 h16* __restrict__ wk16,
                                                 h16* __restrict__ wv16) {
  const int kt = blockIdx.x, head = blockIdx.y, tid = threadIdx.x;
  const size_t tbase = ((size_t)head * KT_ + kt) * 2048;
#pragma unroll
  for (int i = 0; i < 8; ++i) {
    int lin = i * 256 + tid;
    int k = lin >> 6, nn = lin & 63;
    int si = swzB(k, nn);
    size_t go = (size_t)head * E_ * D_ + (size_t)(kt * 32 + k) * D_ + nn;
    wq16[tbase + si] = (h16)Wq[go];
    wk16[tbase + si] = (h16)Wk[go];
    wv16[tbase + si] = (h16)Wv[go];
  }
}

// ------------------------------------------- pre-swizzle lm_W (f16)
// grid: (V/64, KT_).  Tile layout: [nb][kt] so each LM-head block streams
// its 32 K-tiles contiguously.
__global__ __launch_bounds__(256) void k_prepw(const float* __restrict__ W,
                                               h16* __restrict__ w16) {
  const int nb = blockIdx.x, kt = blockIdx.y, tid = threadIdx.x;
  const size_t tbase = ((size_t)nb * KT_ + kt) * 2048;
#pragma unroll
  for (int i = 0; i < 8; ++i) {
    int lin = i * 256 + tid;
    int k = lin >> 6, nn = lin & 63;
    float v = __builtin_nontemporal_load(&W[(size_t)(kt * 32 + k) * V_ +
                                            nb * 64 + nn]);
    w16[tbase + swzB(k, nn)] = (h16)v;
  }
}

// ---------------------------------------------------------------- QKV proj
// grid: (BT/128, H), 256 threads (8 waves), wave = 16 rows x 64 cols.
// Weight tiles arrive via async-to-LDS, double buffered.
// Q stored row-major [B,H,T,D]; K stored in score-B fragment tiles
// (K-dim = head-dim); V stored in 32x64 fragment tiles (K-dim = key).
__global__ __launch_bounds__(256) void k_qkv(const h16* __restrict__ x16,
                                             const h16* __restrict__ wq16,
                                             const h16* __restrict__ wk16,
                                             const h16* __restrict__ wv16,
                                             h16* __restrict__ qo,
                                             h16* __restrict__ kswz,
                                             h16* __restrict__ vswz) {
  __shared__ __align__(16) h16 lbuf[2][3 * 2048];

  const int head  = blockIdx.y;
  const int mbase = blockIdx.x * 128;
  const int tid  = threadIdx.x;
  const int lane = tid & 31, wave = tid >> 5;
  const int g = lane >> 4, n16 = lane & 15;
  const int mrow = mbase + wave * 16 + n16;

  const h16* wq = wq16 + (size_t)head * KT_ * 2048;
  const h16* wk = wk16 + (size_t)head * KT_ * 2048;
  const h16* wv = wv16 + (size_t)head * KT_ * 2048;

  v8f aq[4] = {}, ak[4] = {}, av[4] = {};

  auto issue = [&](int buf, int kt) {
    const size_t o = (size_t)kt * 2048 + tid * 8;   // 16B per thread
    async_cp16(&lbuf[buf][0 * 2048 + tid * 8], wq + o);
    async_cp16(&lbuf[buf][1 * 2048 + tid * 8], wk + o);
    async_cp16(&lbuf[buf][2 * 2048 + tid * 8], wv + o);
  };

  issue(0, 0);
  for (int kt = 0; kt < KT_; ++kt) {
    const int cur = kt & 1;
    if (kt + 1 < KT_) { issue(cur ^ 1, kt + 1); WAIT_ASYNC(3); }
    else              { WAIT_ASYNC(0); }
    __syncthreads();

    Frag16 a;
    const h16* xp = x16 + (size_t)mrow * E_ + kt * 32;
    a.h2[0] = *(const v8h*)(xp + 8 * g);
    a.h2[1] = *(const v8h*)(xp + 16 + 8 * g);

#pragma unroll
    for (int nt = 0; nt < 4; ++nt) {
      const int base = (nt * 32 + lane) << 4;
      Frag16 bq, bk, bv;
      const h16* lq = &lbuf[cur][0];
      bq.h2[0] = *(const v8h*)(lq + base);        bq.h2[1] = *(const v8h*)(lq + base + 8);
      bk.h2[0] = *(const v8h*)(lq + 2048 + base); bk.h2[1] = *(const v8h*)(lq + 2048 + base + 8);
      bv.h2[0] = *(const v8h*)(lq + 4096 + base); bv.h2[1] = *(const v8h*)(lq + 4096 + base + 8);
      aq[nt] = wmma16(a, bq, aq[nt]);
      ak[nt] = wmma16(a, bk, ak[nt]);
      av[nt] = wmma16(a, bv, av[nt]);
    }
    __syncthreads();
  }

#pragma unroll
  for (int nt = 0; nt < 4; ++nt) {
#pragma unroll
    for (int r = 0; r < 8; ++r) {
      const int grow = mbase + wave * 16 + r + 8 * g;
      const int b = grow >> 11, t = grow & (T_ - 1);
      const int d = nt * 16 + n16;
      // Q: row-major
      qo[((size_t)(b * H_ + head) * T_ + t) * D_ + d] = (h16)aq[nt][r];
      // K & V: fragment-ordered 32-key tiles
      const size_t tb = ((size_t)(b * H_ + head) * (T_ / 32) + (t >> 5)) * 2048;
      const int c = (t >> 4) & 1, nk = t & 15;
      const int f = d >> 5;
      const int g2 = n16 >> 3;
      const int h2 = (n16 & 7) | ((nt & 1) << 3);
      kswz[tb + (((c * 2 + f) * 32 + g2 * 16 + nk) << 4) + h2] = (h16)ak[nt][r];
      vswz[tb + swzB(t & 31, d)] = (h16)av[nt][r];
    }
  }
}

// ---------------------------------------------------------------- attention
// grid: B*H*(T/64), 128 threads (4 waves).  Flash attention; K/V 4KB tiles
// arrive via async-to-LDS, double buffered, shared by all 4 waves.
__global__ __launch_bounds__(128) void k_attn(const h16* __restrict__ q16,
                                              const h16* __restrict__ kswz,
                                              const h16* __restrict__ vswz,
                                              h16* __restrict__ o16) {
  __shared__ __align__(16) h16 ldsK[2][2048];
  __shared__ __align__(16) h16 ldsV[2][2048];
  __shared__ __align__(16) h16 ldsp[4][16 * 32];   // P bounce, 1KB / wave

  const int nqc  = T_ / 64;
  const int qc   = blockIdx.x % nqc;
  const int head = (blockIdx.x / nqc) % H_;
  const int b    = blockIdx.x / (nqc * H_);
  const int tid = threadIdx.x, lane = tid & 31, wave = tid >> 5;
  const int g = lane >> 4, n16 = lane & 15;
  const int qbase = qc * 64;
  const int qrow0 = qbase + wave * 16;

  const size_t bh = (size_t)(b * H_ + head);
  const h16* ktiles = kswz + bh * (T_ / 32) * 2048;
  const h16* vtiles = vswz + bh * (T_ / 32) * 2048;

  // Q A-fragments, pre-scaled by D^-1/2 = 0.125 (exact pow2)
  Frag16 qa0, qa1;
  {
    const h16* qp = q16 + (bh * T_ + qrow0 + n16) * D_;
    qa0.h2[0] = *(const v8h*)(qp + 8 * g);
    qa0.h2[1] = *(const v8h*)(qp + 16 + 8 * g);
    qa1.h2[0] = *(const v8h*)(qp + 32 + 8 * g);
    qa1.h2[1] = *(const v8h*)(qp + 48 + 8 * g);
#pragma unroll
    for (int i = 0; i < 16; ++i) {
      qa0.e[i] = qa0.e[i] * (h16)0.125f;
      qa1.e[i] = qa1.e[i] * (h16)0.125f;
    }
  }

  float mrow[8], lrow[8];
  v8f oacc[4] = {};
#pragma unroll
  for (int r = 0; r < 8; ++r) { mrow[r] = -1e30f; lrow[r] = 0.f; }

  h16* lp = &ldsp[wave][0];

  auto issueKV = [&](int buf, int st) {
    const h16* kt = ktiles + (size_t)st * 2048;
    const h16* vt = vtiles + (size_t)st * 2048;
    async_cp16(&ldsK[buf][tid * 8],        kt + tid * 8);         // 128thr*16B
    async_cp16(&ldsK[buf][1024 + tid * 8], kt + 1024 + tid * 8);
    async_cp16(&ldsV[buf][tid * 8],        vt + tid * 8);
    async_cp16(&ldsV[buf][1024 + tid * 8], vt + 1024 + tid * 8);
  };

  const int ntile = (qbase + 64) >> 5;               // block-uniform
  issueKV(0, 0);
  for (int st = 0; st < ntile; ++st) {
    const int cur = st & 1;
    if (st + 1 < ntile) { issueKV(cur ^ 1, st + 1); WAIT_ASYNC(4); }
    else                { WAIT_ASYNC(0); }
    __syncthreads();

    const int s0 = st * 32;
    const h16* lk = &ldsK[cur][0];
    const h16* lv = &ldsV[cur][0];

    // ---- scores: two 16x16 tiles
    v8f sc0, sc1;
#pragma unroll
    for (int c = 0; c < 2; ++c) {
      Frag16 kb0, kb1;
      const int b0 = ((c * 2 + 0) * 32 + lane) << 4;
      const int b1 = ((c * 2 + 1) * 32 + lane) << 4;
      kb0.h2[0] = *(const v8h*)(lk + b0); kb0.h2[1] = *(const v8h*)(lk + b0 + 8);
      kb1.h2[0] = *(const v8h*)(lk + b1); kb1.h2[1] = *(const v8h*)(lk + b1 + 8);
      v8f z = {};
      z = wmma16(qa0, kb0, z);
      z = wmma16(qa1, kb1, z);
      const int key = s0 + c * 16 + n16;
#pragma unroll
      for (int r = 0; r < 8; ++r) {                  // causal mask
        if (key > qrow0 + r + 8 * g) z[r] = -1e30f;
      }
      if (c == 0) sc0 = z; else sc1 = z;
    }

    // ---- online softmax (stats land VGPR-aligned with accumulators)
    float alpha[8], p0a[8], p1a[8];
#pragma unroll
    for (int r = 0; r < 8; ++r) {
      float t = fmaxf(sc0[r], sc1[r]);
      t = fmaxf(t, __shfl_xor(t, 1, 32));
      t = fmaxf(t, __shfl_xor(t, 2, 32));
      t = fmaxf(t, __shfl_xor(t, 4, 32));
      t = fmaxf(t, __shfl_xor(t, 8, 32));
      float nm = fmaxf(mrow[r], t);
      float al = __expf(mrow[r] - nm);
      mrow[r] = nm;
      float e0 = __expf(sc0[r] - nm);
      float e1 = __expf(sc1[r] - nm);
      float rs = e0 + e1;
      rs += __shfl_xor(rs, 1, 32);
      rs += __shfl_xor(rs, 2, 32);
      rs += __shfl_xor(rs, 4, 32);
      rs += __shfl_xor(rs, 8, 32);
      lrow[r] = lrow[r] * al + rs;
      alpha[r] = al; p0a[r] = e0; p1a[r] = e1;
    }
#pragma unroll
    for (int nt = 0; nt < 4; ++nt)
#pragma unroll
      for (int r = 0; r < 8; ++r) oacc[nt][r] *= alpha[r];

    // ---- P: D-layout -> LDS -> A-layout
#pragma unroll
    for (int r = 0; r < 8; ++r) {
      lp[(r + 8 * g) * 32 + n16]      = (h16)p0a[r];
      lp[(r + 8 * g) * 32 + 16 + n16] = (h16)p1a[r];
    }
    __syncthreads();
    Frag16 pa;
    pa.h2[0] = *(const v8h*)(lp + n16 * 32 + 8 * g);
    pa.h2[1] = *(const v8h*)(lp + n16 * 32 + 16 + 8 * g);

    // ---- O += P(16x32) x V(32x64)
#pragma unroll
    for (int nt = 0; nt < 4; ++nt) {
      const int base = (nt * 32 + lane) << 4;
      Frag16 vf;
      vf.h2[0] = *(const v8h*)(lv + base);
      vf.h2[1] = *(const v8h*)(lv + base + 8);
      oacc[nt] = wmma16(pa, vf, oacc[nt]);
    }
    __syncthreads();
  }

  // ---- normalize and store (concat heads)
#pragma unroll
  for (int r = 0; r < 8; ++r) lrow[r] = 1.0f / lrow[r];
#pragma unroll
  for (int nt = 0; nt < 4; ++nt)
#pragma unroll
    for (int r = 0; r < 8; ++r) {
      const int row = qrow0 + r + 8 * g;
      size_t off = (size_t)(b * T_ + row) * E_ + head * D_ + nt * 16 + n16;
      o16[off] = (h16)(oacc[nt][r] * lrow[r]);
    }
}

// ---------------------------------------------------------------- LM head
// grid: (V/64, BT/128), 256 threads (8 waves).  128x64 block tile, K=E.
__global__ __launch_bounds__(256) void k_lmhead(const h16* __restrict__ o16,
                                                const h16* __restrict__ w16,
                                                const float* __restrict__ lmb,
                                                float* __restrict__ out) {
  __shared__ __align__(16) h16 lw[2][2048];

  const int nbase = blockIdx.x * 64;
  const int mbase = blockIdx.y * 128;
  const int tid = threadIdx.x, lane = tid & 31, wave = tid >> 5;
  const int g = lane >> 4, n16 = lane & 15;
  const int mrow = mbase + wave * 16 + n16;
  const h16* wt = w16 + (size_t)blockIdx.x * KT_ * 2048;

  v8f acc[4] = {};

  auto issueW = [&](int buf, int kt) {
    async_cp16(&lw[buf][tid * 8], wt + (size_t)kt * 2048 + tid * 8);
  };

  issueW(0, 0);
  for (int kt = 0; kt < KT_; ++kt) {
    const int cur = kt & 1;
    if (kt + 1 < KT_) { issueW(cur ^ 1, kt + 1); WAIT_ASYNC(1); }
    else              { WAIT_ASYNC(0); }
    __syncthreads();

    Frag16 a;
    const h16* op = o16 + (size_t)mrow * E_ + kt * 32;
    a.h2[0] = *(const v8h*)(op + 8 * g);
    a.h2[1] = *(const v8h*)(op + 16 + 8 * g);

#pragma unroll
    for (int nt = 0; nt < 4; ++nt) {
      const h16* p = &lw[cur][(nt * 32 + lane) << 4];
      Frag16 bw;
      bw.h2[0] = *(const v8h*)(p);
      bw.h2[1] = *(const v8h*)(p + 8);
      acc[nt] = wmma16(a, bw, acc[nt]);
    }
    __syncthreads();
  }

#pragma unroll
  for (int nt = 0; nt < 4; ++nt) {
    const int col = nbase + nt * 16 + n16;
    const float bias = lmb[col];
#pragma unroll
    for (int r = 0; r < 8; ++r) {
      const int grow = mbase + wave * 16 + r + 8 * g;
      __builtin_nontemporal_store(acc[nt][r] + bias,
                                  &out[(size_t)grow * V_ + col]);
    }
  }
}

// ---------------------------------------------------------------- launch
extern "C" void kernel_launch(void* const* d_in, const int* in_sizes, int n_in,
                              void* d_out, int out_size, void* d_ws, size_t ws_size,
                              hipStream_t stream) {
  const int*   X   = (const int*)d_in[0];
  const float* tok = (const float*)d_in[1];
  const float* pos = (const float*)d_in[2];
  const float* Wq  = (const float*)d_in[3];
  const float* Wk  = (const float*)d_in[4];
  const float* Wv  = (const float*)d_in[5];
  const float* lmW = (const float*)d_in[6];
  const float* lmb = (const float*)d_in[7];
  float* out = (float*)d_out;

  constexpr size_t SZ  = (size_t)BT_ * E_;           // 4.19M halves
  constexpr size_t WSZ = (size_t)H_ * KT_ * 2048;    // 1.05M halves
  h16* x16  = (h16*)d_ws;
  h16* q16  = x16 + SZ;
  h16* ksw  = q16 + SZ;
  h16* vsw  = ksw + SZ;
  h16* o16  = vsw + SZ;
  h16* wq16 = o16 + SZ;
  h16* wk16 = wq16 + WSZ;
  h16* wv16 = wk16 + WSZ;
  h16* w16  = wv16 + WSZ;                            // (V/64)*KT_*2048 halves

  k_embed<<<BT_, 256, 0, stream>>>(X, tok, pos, x16);

  dim3 gpq(KT_, H_);
  k_prepqkv<<<gpq, 256, 0, stream>>>(Wq, Wk, Wv, wq16, wk16, wv16);

  dim3 gpw(V_ / 64, KT_);
  k_prepw<<<gpw, 256, 0, stream>>>(lmW, w16);

  dim3 gq(BT_ / 128, H_);
  k_qkv<<<gq, 256, 0, stream>>>(x16, wq16, wk16, wv16, q16, ksw, vsw);

  k_attn<<<B_ * H_ * (T_ / 64), 128, 0, stream>>>(q16, ksw, vsw, o16);

  dim3 gl(V_ / 64, BT_ / 128);
  k_lmhead<<<gl, 256, 0, stream>>>(o16, w16, lmb, out);
}
